// MambaRegressor_30382598652285
// MI455X (gfx1250) — matrix operand order repaired and checked by reading, not compiled, not run on hardware
//
#include <hip/hip_runtime.h>

#define B_SZ 32
#define NPAT 512
#define PSZ  16
#define DM   1024
#define DSN  16
#define DC   4
#define DI   2048
#define DTR  64
#define NTOK (B_SZ * NPAT)     /* 16384 rows */
#define XDW  (DTR + 2 * DSN)   /* 96 */

typedef __attribute__((ext_vector_type(16))) __bf16       v16bf;
typedef __attribute__((ext_vector_type(8)))  float        v8f;
typedef __attribute__((ext_vector_type(4)))  unsigned int u32x4;
typedef __attribute__((ext_vector_type(4)))  int          i32x4;

#ifndef __has_builtin
#define __has_builtin(x) 0
#endif
#if __has_builtin(__builtin_amdgcn_global_load_async_to_lds_b128) && \
    __has_builtin(__builtin_amdgcn_s_wait_asynccnt)
#define USE_ASYNC_LDS 1
#else
#define USE_ASYNC_LDS 0
#endif

__device__ __forceinline__ unsigned short f2bf(float f) {
  unsigned u = __float_as_uint(f);
  unsigned r = ((u >> 16) & 1u) + 0x7FFFu;   // round-to-nearest-even
  return (unsigned short)((u + r) >> 16);
}
__device__ __forceinline__ float silu_(float x) { return x / (1.f + __expf(-x)); }

#if USE_ASYNC_LDS
// 16B contiguous global -> LDS async copy (GLOBAL_LOAD_ASYNC_TO_LDS_B128).
// Builtin takes generic i32x4* params (per compiler diagnostic); src, dst order.
__device__ __forceinline__ void cp_async16(const void* g, void* l) {
  __builtin_amdgcn_global_load_async_to_lds_b128((i32x4*)g, (i32x4*)l, 0, 0);
}
#endif

// ------------------------------------------ weight convert + transpose to [N,K]
__global__ __launch_bounds__(256)
void k_w_to_bf16_t(const float* __restrict__ W,          // [K,N] f32
                   unsigned short* __restrict__ Wt,      // [N,K] bf16 bits
                   int K, int N) {
  int o = blockIdx.x * 256 + threadIdx.x;
  if (o < K * N) {
    int n = o / K, k = o - n * K;
    Wt[o] = f2bf(W[(size_t)k * N + n]);
  }
}

// ------------------------------------------------- block reduction helper
__device__ __forceinline__ float block_sum(float v, float* red, int tid) {
  red[tid] = v; __syncthreads();
  for (int off = 128; off > 0; off >>= 1) {
    if (tid < off) red[tid] += red[tid + off];
    __syncthreads();
  }
  float r = red[0];
  __syncthreads();
  return r;
}

// ---------------------------------------- patch embed: LN -> Wp -> LN -> +pos
__global__ __launch_bounds__(256)
void k_patch_embed(const float* __restrict__ x,
                   const float* __restrict__ ln1_g, const float* __restrict__ ln1_b,
                   const float* __restrict__ Wp,    const float* __restrict__ bp,
                   const float* __restrict__ ln2_g, const float* __restrict__ ln2_b,
                   unsigned short* __restrict__ h_bf) {
  const int row = blockIdx.x;          // b*NPAT + n
  const int n   = row & (NPAT - 1);
  const int tid = threadIdx.x;
  __shared__ float pn[PSZ];
  __shared__ float red[256];
  if (tid < PSZ) pn[tid] = x[(size_t)row * PSZ + tid];
  __syncthreads();
  float m = 0.f;
  #pragma unroll
  for (int k = 0; k < PSZ; ++k) m += pn[k];
  m *= (1.f / PSZ);
  float v = 0.f;
  #pragma unroll
  for (int k = 0; k < PSZ; ++k) { float d = pn[k] - m; v += d * d; }
  v *= (1.f / PSZ);
  float rs = rsqrtf(v + 1e-5f);
  float p[PSZ];
  #pragma unroll
  for (int k = 0; k < PSZ; ++k) p[k] = (pn[k] - m) * rs * ln1_g[k] + ln1_b[k];
  float o[4];
  #pragma unroll
  for (int i = 0; i < 4; ++i) {
    int j = tid + 256 * i;
    float acc = bp[j];
    #pragma unroll
    for (int k = 0; k < PSZ; ++k) acc += p[k] * Wp[k * DM + j];
    o[i] = acc;
  }
  float s = block_sum(o[0] + o[1] + o[2] + o[3], red, tid);
  float mean = s * (1.f / DM);
  float sq = 0.f;
  #pragma unroll
  for (int i = 0; i < 4; ++i) { float d = o[i] - mean; sq += d * d; }
  float var = block_sum(sq, red, tid) * (1.f / DM);
  float rs2 = rsqrtf(var + 1e-5f);
  #pragma unroll
  for (int i = 0; i < 4; ++i) {
    int j = tid + 256 * i;
    float val = (o[i] - mean) * rs2 * ln2_g[j] + ln2_b[j];
    int jj = (j < DM / 2) ? j : j - DM / 2;
    float omega = __expf(-(float)jj * (9.210340371976184f / 511.0f)); // 1e4^(-jj/511)
    float ang = (float)n * omega;
    val += (j < DM / 2) ? __sinf(ang) : __cosf(ang);
    h_bf[(size_t)row * DM + j] = f2bf(val);
  }
}

// --------------------------------------------------- bf16 WMMA GEMM, C = A*Wt^T
// A [M,K], Wt [N,K] (pre-transposed weights), C [M,N] f32.
// Block tile 128x128, BK=32; 8 waves in 4(M)x2(N) grid; each wave 2x4 WMMA tiles.
#define LDSS 40   // ushort row stride: 80B -> 16B aligned, bank-spreading
__device__ __forceinline__ void wmma_tile_step(const unsigned short* As,
                                               const unsigned short* Bs,
                                               int wm, int wn, int hf, int l15,
                                               v8f (&acc)[2][4]) {
  union Frag { u32x4 q[2]; v16bf v; } af[2], bfr[4];
  #pragma unroll
  for (int i = 0; i < 2; ++i) {                 // A frag: K = hf*8+{0..7}, +16..
    int r = wm * 32 + i * 16 + l15;
    af[i].q[0] = *(const u32x4*)&As[r * LDSS + hf * 8];
    af[i].q[1] = *(const u32x4*)&As[r * LDSS + hf * 8 + 16];
  }
  #pragma unroll
  for (int j = 0; j < 4; ++j) {                 // B frag: K = hf*8+{0..7}, +16..
    int c = wn * 64 + j * 16 + l15;
    bfr[j].q[0] = *(const u32x4*)&Bs[c * LDSS + hf * 8];
    bfr[j].q[1] = *(const u32x4*)&Bs[c * LDSS + hf * 8 + 16];
  }
  #pragma unroll
  for (int i = 0; i < 2; ++i)
    #pragma unroll
    for (int j = 0; j < 4; ++j)
      acc[i][j] = __builtin_amdgcn_wmma_f32_16x16x32_bf16(
          false, af[i].v, false, bfr[j].v, (short)0, acc[i][j], false, false);
}

__global__ __launch_bounds__(256)
void k_gemm_bf16(const unsigned short* __restrict__ A,   // [M,K] bf16 bits
                 const unsigned short* __restrict__ Wt,  // [N,K] bf16 bits
                 float* __restrict__ C,                   // [M,N] f32
                 int M, int N, int K) {
  __shared__ __align__(16) unsigned short As[2][128 * LDSS];
  __shared__ __align__(16) unsigned short Bs[2][128 * LDSS];
  const int tid  = threadIdx.x;
  const int lane = tid & 31, wave = tid >> 5;
  const int wm = wave & 3, wn = wave >> 2;
  const int hf = lane >> 4, l15 = lane & 15;
  const int m0 = blockIdx.y * 128, n0 = blockIdx.x * 128;
  // tile staging: 128 rows x 32 k, 2 x (16B = 8 bf16) chunks per thread
  const int r0 = tid >> 2, r1 = r0 + 64, c8 = (tid & 3) * 8;
  const int bn0 = min(n0 + r0, N - 1);           // clamp keeps loads in-bounds
  const int bn1 = min(n0 + r1, N - 1);
  v8f acc[2][4] = {};

#if USE_ASYNC_LDS
  // prologue: stage tiles 0 and 1 (4 async issues per thread per tile)
  cp_async16(&A[(size_t)(m0 + r0) * K + c8],  &As[0][r0 * LDSS + c8]);
  cp_async16(&A[(size_t)(m0 + r1) * K + c8],  &As[0][r1 * LDSS + c8]);
  cp_async16(&Wt[(size_t)bn0 * K + c8],       &Bs[0][r0 * LDSS + c8]);
  cp_async16(&Wt[(size_t)bn1 * K + c8],       &Bs[0][r1 * LDSS + c8]);
  if (K > 32) {
    cp_async16(&A[(size_t)(m0 + r0) * K + 32 + c8], &As[1][r0 * LDSS + c8]);
    cp_async16(&A[(size_t)(m0 + r1) * K + 32 + c8], &As[1][r1 * LDSS + c8]);
    cp_async16(&Wt[(size_t)bn0 * K + 32 + c8],      &Bs[1][r0 * LDSS + c8]);
    cp_async16(&Wt[(size_t)bn1 * K + 32 + c8],      &Bs[1][r1 * LDSS + c8]);
  }
  for (int k0 = 0; k0 < K; k0 += 32) {
    const int cur = (k0 >> 5) & 1;
    // async loads retire in order: <=4 leaves only the next tile in flight
    if (k0 + 32 < K) __builtin_amdgcn_s_wait_asynccnt(4);
    else             __builtin_amdgcn_s_wait_asynccnt(0);
    __syncthreads();                              // all waves see current buffer
    wmma_tile_step(As[cur], Bs[cur], wm, wn, hf, l15, acc);
    __syncthreads();                              // all reads of cur done
    if (k0 + 64 < K) {
      const int kn = k0 + 64;
      cp_async16(&A[(size_t)(m0 + r0) * K + kn + c8], &As[cur][r0 * LDSS + c8]);
      cp_async16(&A[(size_t)(m0 + r1) * K + kn + c8], &As[cur][r1 * LDSS + c8]);
      cp_async16(&Wt[(size_t)bn0 * K + kn + c8],      &Bs[cur][r0 * LDSS + c8]);
      cp_async16(&Wt[(size_t)bn1 * K + kn + c8],      &Bs[cur][r1 * LDSS + c8]);
    }
  }
#else
  for (int k0 = 0; k0 < K; k0 += 32) {
    u32x4 a0 = *(const u32x4*)&A[(size_t)(m0 + r0) * K + k0 + c8];
    u32x4 a1 = *(const u32x4*)&A[(size_t)(m0 + r1) * K + k0 + c8];
    u32x4 b0 = *(const u32x4*)&Wt[(size_t)bn0 * K + k0 + c8];
    u32x4 b1 = *(const u32x4*)&Wt[(size_t)bn1 * K + k0 + c8];
    __syncthreads();
    *(u32x4*)&As[0][r0 * LDSS + c8] = a0;
    *(u32x4*)&As[0][r1 * LDSS + c8] = a1;
    *(u32x4*)&Bs[0][r0 * LDSS + c8] = b0;
    *(u32x4*)&Bs[0][r1 * LDSS + c8] = b1;
    __syncthreads();
    wmma_tile_step(As[0], Bs[0], wm, wn, hf, l15, acc);
  }
#endif

  #pragma unroll
  for (int i = 0; i < 2; ++i) {
    int r = m0 + wm * 32 + i * 16 + hf * 8;       // C/D: rows hf*8+q, col l15
    #pragma unroll
    for (int j = 0; j < 4; ++j) {
      int c = n0 + wn * 64 + j * 16 + l15;
      if (c < N) {
        #pragma unroll
        for (int q = 0; q < 8; ++q)
          C[(size_t)(r + q) * N + c] = acc[i][j][q];
      }
    }
  }
}

// -------------------------------------- causal depthwise conv (DC=4) + SiLU
__global__ __launch_bounds__(256)
void k_conv_silu(const float* __restrict__ xz, const float* __restrict__ conv_w,
                 const float* __restrict__ conv_b,
                 float* __restrict__ uc, unsigned short* __restrict__ uc_bf) {
  size_t idx = (size_t)blockIdx.x * 256 + threadIdx.x;   // over NTOK*DI
  int d = (int)(idx & (DI - 1));
  int row = (int)(idx >> 11);
  int t = row & (NPAT - 1);
  int b = row >> 9;
  float acc = conv_b[d];
  #pragma unroll
  for (int k = 0; k < DC; ++k) {
    int ts = t - (DC - 1) + k;
    if (ts >= 0)
      acc += xz[((size_t)(b * NPAT + ts)) * (2 * DI) + d] * conv_w[d * DC + k];
  }
  float s = silu_(acc);
  uc[idx] = s;
  uc_bf[idx] = f2bf(s);
}

// ----------------------------------------------------- dt slice -> bf16 rows
__global__ __launch_bounds__(256)
void k_extract_dt(const float* __restrict__ xd, unsigned short* __restrict__ dt_bf) {
  int i = blockIdx.x * 256 + threadIdx.x;                // over NTOK*DTR
  int row = i >> 6, c = i & 63;
  dt_bf[i] = f2bf(xd[(size_t)row * XDW + c]);
}

// ------------------------------------------------------ selective-state scan
__global__ __launch_bounds__(256)
void k_ssm_scan(const float* __restrict__ dtW,           // [NTOK, DI] pre-softplus
                const float* __restrict__ xd,            // [NTOK, 96] (B,C in 64..95)
                const float* __restrict__ uc,            // [NTOK, DI]
                const float* __restrict__ A_log, const float* __restrict__ b_dt,
                const float* __restrict__ Dv,
                float* __restrict__ ys) {                // may alias dtW
  const int tid = threadIdx.x;
  const int d = blockIdx.x * 256 + tid;
  const int b = blockIdx.y;
  __shared__ float BC[2][2 * DSN];
  float Ar[DSN];
  #pragma unroll
  for (int s = 0; s < DSN; ++s) Ar[s] = -__expf(A_log[(size_t)d * DSN + s]);
  const float bdt = b_dt[d];
  const float Dd = Dv[d];
  float h[DSN];
  #pragma unroll
  for (int s = 0; s < DSN; ++s) h[s] = 0.f;
  for (int t = 0; t < NPAT; ++t) {
    const size_t row = (size_t)b * NPAT + t;
    if (tid < 2 * DSN) BC[t & 1][tid] = xd[row * XDW + DTR + tid];
    __syncthreads();
    const float* bc = BC[t & 1];
    float x0 = dtW[row * DI + d] + bdt;
    float delta = (x0 > 20.f) ? x0 : __logf(1.f + __expf(x0));   // softplus
    float u = uc[row * DI + d];
    float du = delta * u;
    float y = 0.f;
    #pragma unroll
    for (int s = 0; s < DSN; ++s) {
      float dA = __expf(delta * Ar[s]);
      h[s] = h[s] * dA + du * bc[s];
      y += h[s] * bc[DSN + s];
    }
    ys[row * DI + d] = y + u * Dd;
  }
}

// -------------------------------------------------------------- gate by z
__global__ __launch_bounds__(256)
void k_gate(const float* __restrict__ ys, const float* __restrict__ xz,
            unsigned short* __restrict__ yg_bf) {
  size_t idx = (size_t)blockIdx.x * 256 + threadIdx.x;   // over NTOK*DI
  int d = (int)(idx & (DI - 1));
  size_t row = idx >> 11;
  float z = xz[row * (2 * DI) + DI + d];
  yg_bf[idx] = f2bf(ys[idx] * silu_(z));
}

// ------------------------------------------- mean-pool + LN + linear head
__global__ __launch_bounds__(256)
void k_pool_head(const float* __restrict__ yout, const float* __restrict__ ln3_g,
                 const float* __restrict__ ln3_b, const float* __restrict__ Wh,
                 const float* __restrict__ bh, float* __restrict__ out) {
  const int b = blockIdx.x, tid = threadIdx.x;
  __shared__ float red[256];
  float acc[4] = {0.f, 0.f, 0.f, 0.f};
  for (int t = 0; t < NPAT; ++t) {
    const float* rp = &yout[((size_t)b * NPAT + t) * DM];
    #pragma unroll
    for (int i = 0; i < 4; ++i) acc[i] += rp[tid + 256 * i];
  }
  #pragma unroll
  for (int i = 0; i < 4; ++i) acc[i] *= (1.f / NPAT);
  float mean = block_sum(acc[0] + acc[1] + acc[2] + acc[3], red, tid) * (1.f / DM);
  float sq = 0.f;
  #pragma unroll
  for (int i = 0; i < 4; ++i) { float dd = acc[i] - mean; sq += dd * dd; }
  float var = block_sum(sq, red, tid) * (1.f / DM);
  float rs = rsqrtf(var + 1e-5f);
  float pd = 0.f;
  #pragma unroll
  for (int i = 0; i < 4; ++i) {
    int j = tid + 256 * i;
    pd += ((acc[i] - mean) * rs * ln3_g[j] + ln3_b[j]) * Wh[j];
  }
  float dot = block_sum(pd, red, tid);
  if (tid == 0) out[b] = dot + bh[0];
}

// ============================================================== launcher
extern "C" void kernel_launch(void* const* d_in, const int* in_sizes, int n_in,
                              void* d_out, int out_size, void* d_ws, size_t ws_size,
                              hipStream_t stream) {
  const float* x      = (const float*)d_in[0];
  const float* ln1_g  = (const float*)d_in[1];
  const float* ln1_b  = (const float*)d_in[2];
  const float* Wp     = (const float*)d_in[3];
  const float* bp     = (const float*)d_in[4];
  const float* ln2_g  = (const float*)d_in[5];
  const float* ln2_b  = (const float*)d_in[6];
  const float* W_in   = (const float*)d_in[7];
  const float* conv_w = (const float*)d_in[8];
  const float* conv_b = (const float*)d_in[9];
  const float* W_x    = (const float*)d_in[10];
  const float* W_dt   = (const float*)d_in[11];
  const float* b_dt   = (const float*)d_in[12];
  const float* A_log  = (const float*)d_in[13];
  const float* Dv     = (const float*)d_in[14];
  const float* W_out  = (const float*)d_in[15];
  const float* ln3_g  = (const float*)d_in[16];
  const float* ln3_b  = (const float*)d_in[17];
  const float* Wh     = (const float*)d_in[18];
  const float* bh     = (const float*)d_in[19];
  float* out = (float*)d_out;

  char* ws = (char*)d_ws;
  size_t off = 0;
  auto take = [&](size_t bytes) -> char* {
    char* p = ws + off;
    off = (off + bytes + 255) & ~(size_t)255;
    return p;
  };
  unsigned short* h_bf    = (unsigned short*)take((size_t)NTOK * DM * 2);
  unsigned short* Win_t   = (unsigned short*)take((size_t)DM * 2 * DI * 2);  // [4096,1024]
  unsigned short* Wx_t    = (unsigned short*)take((size_t)DI * XDW * 2);     // [96,2048]
  unsigned short* Wdt_t   = (unsigned short*)take((size_t)DTR * DI * 2);     // [2048,64]
  unsigned short* Wout_t  = (unsigned short*)take((size_t)DI * DM * 2);      // [1024,2048]
  float*          xz      = (float*)take((size_t)NTOK * 2 * DI * 4);
  float*          uc      = (float*)take((size_t)NTOK * DI * 4);
  unsigned short* uc_bf   = (unsigned short*)take((size_t)NTOK * DI * 2);
  float*          xd      = (float*)take((size_t)NTOK * XDW * 4);
  unsigned short* dt_bf   = (unsigned short*)take((size_t)NTOK * DTR * 2);
  float*          dtW     = (float*)take((size_t)NTOK * DI * 4);
  // aliases (lifetimes disjoint / element-wise read-then-write):
  float*          ysb     = dtW;                       // scan output overwrites dtW
  unsigned short* yg_bf   = uc_bf;                     // gate output reuses uc_bf
  float*          yout    = xz;                        // out_proj result reuses xz

  // 1) weights -> bf16, transposed to [N,K] (enables straight-copy LDS staging)
  {
    int n1 = DM * 2 * DI, n2 = DI * XDW, n3 = DTR * DI, n4 = DI * DM;
    k_w_to_bf16_t<<<dim3((n1 + 255) / 256), dim3(256), 0, stream>>>(W_in, Win_t, DM, 2 * DI);
    k_w_to_bf16_t<<<dim3((n2 + 255) / 256), dim3(256), 0, stream>>>(W_x, Wx_t, DI, XDW);
    k_w_to_bf16_t<<<dim3((n3 + 255) / 256), dim3(256), 0, stream>>>(W_dt, Wdt_t, DTR, DI);
    k_w_to_bf16_t<<<dim3((n4 + 255) / 256), dim3(256), 0, stream>>>(W_out, Wout_t, DI, DM);
  }
  // 2) patch embedding (fused LN->Wp->LN->posemb), bf16 out
  k_patch_embed<<<dim3(NTOK), dim3(256), 0, stream>>>(x, ln1_g, ln1_b, Wp, bp,
                                                      ln2_g, ln2_b, h_bf);
  // 3) xz = h @ W_in                [16384,1024]x[1024,4096]
  k_gemm_bf16<<<dim3((2 * DI) / 128, NTOK / 128), dim3(256), 0, stream>>>(
      h_bf, Win_t, xz, NTOK, 2 * DI, DM);
  // 4) uc = silu(causal depthwise conv(u))
  k_conv_silu<<<dim3((NTOK * DI) / 256), dim3(256), 0, stream>>>(xz, conv_w, conv_b,
                                                                 uc, uc_bf);
  // 5) xd = uc @ W_x                [16384,2048]x[2048,96]
  k_gemm_bf16<<<dim3((XDW + 127) / 128, NTOK / 128), dim3(256), 0, stream>>>(
      uc_bf, Wx_t, xd, NTOK, XDW, DI);
  // 6) dt slice -> bf16
  k_extract_dt<<<dim3((NTOK * DTR) / 256), dim3(256), 0, stream>>>(xd, dt_bf);
  // 7) dtW = dt @ W_dt              [16384,64]x[64,2048]
  k_gemm_bf16<<<dim3(DI / 128, NTOK / 128), dim3(256), 0, stream>>>(
      dt_bf, Wdt_t, dtW, NTOK, DI, DTR);
  // 8) selective scan (softplus fused), ys overwrites dtW element-wise
  k_ssm_scan<<<dim3(DI / 256, B_SZ), dim3(256), 0, stream>>>(dtW, xd, uc, A_log,
                                                             b_dt, Dv, ysb);
  // 9) gate by silu(z)
  k_gate<<<dim3((NTOK * DI) / 256), dim3(256), 0, stream>>>(ysb, xz, yg_bf);
  // 10) yout = ygate @ W_out        [16384,2048]x[2048,1024]  (reuses xz)
  k_gemm_bf16<<<dim3(DM / 128, NTOK / 128), dim3(256), 0, stream>>>(
      yg_bf, Wout_t, yout, NTOK, DM, DI);
  // 11) mean-pool + LN + head
  k_pool_head<<<dim3(B_SZ), dim3(256), 0, stream>>>(yout, ln3_g, ln3_b, Wh, bh, out);
}
